// Attention_42253888258894
// MI455X (gfx1250) — matrix operand order repaired and checked by reading, not compiled, not run on hardware
//
#include <hip/hip_runtime.h>
#include <hip/hip_bf16.h>

// ---------------------------------------------------------------------------
// MI455X (gfx1250) fused attention layer: QKV GEMM -> flash attention -> proj.
// Matmuls: v_wmma_f32_16x16x32_bf16 (wave32). K/V tiles move via the Tensor
// Data Mover (tensor_load_to_lds + s_wait_tensorcnt), double-buffered in LDS.
// ---------------------------------------------------------------------------

#define DIM    512
#define S_LEN  4096
#define NH     8
#define HD     64
#define N_QKV  1536
#define KC     64          // keys per attention chunk
static constexpr float SCALE = 0.125f; // 1/sqrt(64)

typedef __attribute__((ext_vector_type(16))) __bf16        v16bf;
typedef __attribute__((ext_vector_type(8)))  float         v8f;
typedef __attribute__((ext_vector_type(8)))  unsigned int  v8u;
typedef __attribute__((ext_vector_type(4)))  unsigned int  v4u;
typedef __attribute__((ext_vector_type(4)))  unsigned int  u32x4;
typedef __attribute__((ext_vector_type(8)))  int           i32x8;
typedef __attribute__((ext_vector_type(4)))  int           i32x4;

union FragAB {
    v8u   u;
    v16bf bf;
    v4u   u4[2];
};

// f32 -> bf16 round-to-nearest-even (bit trick)
__device__ __forceinline__ unsigned short f2bf(float f) {
    unsigned int u = __float_as_uint(f);
    u += 0x7FFFu + ((u >> 16) & 1u);
    return (unsigned short)(u >> 16);
}

__device__ __forceinline__ v8f wmma_bf16(const FragAB& a, const FragAB& b, v8f c) {
    // 8 args: (neg_a, A, neg_b, B, c_mod, C, reuse_a, reuse_b)
    return __builtin_amdgcn_wmma_f32_16x16x32_bf16(
        false, a.bf, false, b.bf, (short)0, c, false, false);
}

// reductions across one 16-lane half (rows live in halves; masks<16 stay inside)
__device__ __forceinline__ float rowmax16(float v) {
    v = fmaxf(v, __shfl_xor(v, 1, 32));
    v = fmaxf(v, __shfl_xor(v, 2, 32));
    v = fmaxf(v, __shfl_xor(v, 4, 32));
    v = fmaxf(v, __shfl_xor(v, 8, 32));
    return v;
}
__device__ __forceinline__ float rowsum16(float v) {
    v += __shfl_xor(v, 1, 32);
    v += __shfl_xor(v, 2, 32);
    v += __shfl_xor(v, 4, 32);
    v += __shfl_xor(v, 8, 32);
    return v;
}

// ---------------------------------------------------------------------------
// TDM: 2-D tile load, global -> LDS.  Builds the D# descriptor per ISA ch.8:
//  group0: count=1 | lds_addr | global_addr(57b) | type=2
//  group1: data_size=2B, tensor_dim0/1, tile_dim0/1, tensor_dim0_stride
// Issued once per wave (EXEC ignored); tracked by TENSORcnt.
// This toolchain exposes the 6-arg builtin: (g0, g1, g2, g3, g4, cpol);
// groups 2/3/4 are unused for a 2-D tile -> zero-filled.
// ---------------------------------------------------------------------------
__device__ __forceinline__ void tdm_load_2d(
    unsigned lds_addr, const void* gptr,
    unsigned tensor_d0, unsigned tensor_d1, unsigned stride0,
    unsigned tile_d0, unsigned tile_d1)
{
    unsigned long long ga = (unsigned long long)gptr;
    u32x4 g0;
    g0[0] = 1u;                                            // count=1, user desc
    g0[1] = lds_addr;                                      // LDS byte offset
    g0[2] = (unsigned)ga;                                  // global_addr[31:0]
    g0[3] = (unsigned)((ga >> 32) & 0x01FFFFFFu)           // global_addr[56:32]
          | 0x80000000u;                                   // type=2 ("image")
    i32x8 g1;
    g1[0] = (int)(1u << 16);                               // data_size=1 -> 2B
    g1[1] = (int)((tensor_d0 & 0xFFFFu) << 16);            // tensor_dim0[15:0]
    g1[2] = (int)(((tensor_d0 >> 16) & 0xFFFFu)
          |        ((tensor_d1 & 0xFFFFu) << 16));         // dim0 hi | dim1 lo
    g1[3] = (int)(((tensor_d1 >> 16) & 0xFFFFu)
          |        ((tile_d0 & 0xFFFFu) << 16));           // dim1 hi | tile0
    g1[4] = (int)(tile_d1 & 0xFFFFu);                      // tile1 | tile2=0
    g1[5] = (int)stride0;                                  // dim0_stride[31:0]
    g1[6] = 0;                                             // stride hi / dim1_stride
    g1[7] = 0;
    i32x4 z4 = {0, 0, 0, 0};                               // groups 2/3 unused (2-D)
    i32x8 z8 = {0, 0, 0, 0, 0, 0, 0, 0};                   // extra group unused
    __builtin_amdgcn_tensor_load_to_lds(g0, g1, z4, z4, z8, 0);
}

__device__ __forceinline__ unsigned lds_off(const void* p) {
    return (unsigned)(unsigned long long)(uintptr_t)p;     // low 32 = LDS offset
}

// ---------------------------------------------------------------------------
// Kernel 1: qkv = x @ W_qkv + b_qkv. Q,K -> [B,H,S,Hd] bf16; V -> [B,H,Hd,S]
// (pre-transposed so the attention kernel's PV B-matrix is a plain 2-D tile).
// ---------------------------------------------------------------------------
__global__ __launch_bounds__(128) void qkv_kernel(
    const float* __restrict__ x, const float* __restrict__ Wqkv,
    const float* __restrict__ bqkv,
    unsigned short* __restrict__ Q, unsigned short* __restrict__ K,
    unsigned short* __restrict__ V)
{
    __shared__ unsigned short Ash[64 * 40];   // x tile, row-major (A layout)
    __shared__ unsigned short Bsh[64 * 40];   // W tile, transposed (Bt layout)

    const int tid  = threadIdx.x;
    const int wave = tid >> 5;
    const int lane = tid & 31;
    const int l    = lane & 15;
    const int h    = lane >> 4;
    const int mBase = blockIdx.x * 64;        // over B*S = 8192 rows
    const int nBase = blockIdx.y * 64;        // over 3*DIM = 1536 cols

    v8f c[4] = {};

    for (int kt = 0; kt < DIM; kt += 32) {
#pragma unroll
        for (int i = 0; i < 16; ++i) {        // 64x32 A tile
            int idx = tid + i * 128;
            int r = idx >> 5, cc = idx & 31;
            Ash[r * 40 + cc] = f2bf(x[(size_t)(mBase + r) * DIM + kt + cc]);
        }
#pragma unroll
        for (int i = 0; i < 16; ++i) {        // 32x64 W tile -> transposed
            int idx = tid + i * 128;
            int kk = idx >> 6, nn = idx & 63;
            Bsh[nn * 40 + kk] = f2bf(Wqkv[(size_t)(kt + kk) * N_QKV + nBase + nn]);
        }
        __syncthreads();

        FragAB a;
        const unsigned short* ap = &Ash[(wave * 16 + l) * 40];
        a.u4[0] = *(const v4u*)(ap + 8 * h);
        a.u4[1] = *(const v4u*)(ap + 16 + 8 * h);
#pragma unroll
        for (int j = 0; j < 4; ++j) {
            FragAB bfr;
            const unsigned short* bp = &Bsh[(j * 16 + l) * 40 + 8 * h];
            bfr.u4[0] = *(const v4u*)(bp);
            bfr.u4[1] = *(const v4u*)(bp + 16);
            c[j] = wmma_bf16(a, bfr, c[j]);
        }
        __syncthreads();
    }

#pragma unroll
    for (int j = 0; j < 4; ++j) {
        int ncol  = nBase + j * 16 + l;
        float bias = bqkv[ncol];
        int which = ncol / DIM;               // 0=q 1=k 2=v
        int hh = (ncol % DIM) / HD;
        int d  = ncol % HD;
        unsigned short* dst = (which == 0) ? Q : (which == 1) ? K : V;
#pragma unroll
        for (int r = 0; r < 8; ++r) {
            int row = mBase + wave * 16 + r + 8 * h;     // 0..8191
            int b   = row >> 12;
            int sq  = row & (S_LEN - 1);
            size_t off = (which == 2)
                ? ((size_t)(b * NH + hh) * HD + d) * S_LEN + sq       // V^T
                : ((size_t)(b * NH + hh) * S_LEN + sq) * HD + d;      // Q,K
            dst[off] = f2bf(c[j][r] + bias);
        }
    }
}

// ---------------------------------------------------------------------------
// Kernel 2: flash attention. 4 waves x 16 queries; 64-key chunks streamed via
// TDM into double-buffered LDS tiles. 16 WMMA per chunk.
// ---------------------------------------------------------------------------
__global__ __launch_bounds__(128) void attn_kernel(
    const unsigned short* __restrict__ Q, const unsigned short* __restrict__ K,
    const unsigned short* __restrict__ Vt, unsigned short* __restrict__ O)
{
    __shared__ unsigned short Ksh[2][KC * HD];   // K chunk  [t][d] (Bt for Q*K^T)
    __shared__ unsigned short Vsh[2][HD * KC];   // Vt chunk [d][t] (Bt for P*V)
    __shared__ unsigned short Psh[4][16 * KC];   // per-wave P bounce (C->A)

    const int tid  = threadIdx.x;
    const int wave = tid >> 5;
    const int lane = tid & 31;
    const int l    = lane & 15;
    const int h    = lane >> 4;
    const int head = blockIdx.y;
    const int b    = blockIdx.z;
    const size_t base = (size_t)(b * NH + head) * S_LEN * HD;
    const unsigned short* Kp  = K  + base;           // [S,Hd] row-major
    const unsigned short* Vtp = Vt + base;           // [Hd,S] row-major
    const int qRow = blockIdx.x * 64 + wave * 16;

    // resident Q fragments: 16 rows x 64 dims = 2 A-frags
    FragAB qf[2];
    const unsigned short* qp = Q + base + (size_t)(qRow + l) * HD;
#pragma unroll
    for (int ks = 0; ks < 2; ++ks) {
        qf[ks].u4[0] = *(const v4u*)(qp + ks * 32 + 8 * h);
        qf[ks].u4[1] = *(const v4u*)(qp + ks * 32 + 16 + 8 * h);
    }

    float m_i[8], l_i[8], alpha[8];
    v8f acc[4] = {};
#pragma unroll
    for (int r = 0; r < 8; ++r) { m_i[r] = -INFINITY; l_i[r] = 0.f; }

    const int NCHUNK = S_LEN / KC;

    // prologue: TDM-load chunk 0 into buffer 0 (wave 0 owns the TENSORcnt)
    if (wave == 0) {
        tdm_load_2d(lds_off(&Ksh[0][0]), Kp,        HD,    S_LEN, HD,    HD, KC);
        tdm_load_2d(lds_off(&Vsh[0][0]), Vtp,       S_LEN, HD,    S_LEN, KC, HD);
    }

    for (int ic = 0; ic < NCHUNK; ++ic) {
        const int buf = ic & 1;
        if (wave == 0) __builtin_amdgcn_s_wait_tensorcnt(0);
        __syncthreads();   // tiles visible; everyone done with buf^1's old data

        if (wave == 0 && ic + 1 < NCHUNK) {   // prefetch next chunk via TDM
            int kc2 = (ic + 1) * KC;
            tdm_load_2d(lds_off(&Ksh[buf ^ 1][0]), Kp + (size_t)kc2 * HD,
                        HD,    S_LEN, HD,    HD, KC);
            tdm_load_2d(lds_off(&Vsh[buf ^ 1][0]), Vtp + kc2,
                        S_LEN, HD,    S_LEN, KC, HD);
        }

        const unsigned short* Kc = Ksh[buf];
        const unsigned short* Vc = Vsh[buf];

        // scores: S16x64 = Q(16x64) x K^T(64x64) -> 8 WMMA
        v8f sc[4] = {};
#pragma unroll
        for (int nsub = 0; nsub < 4; ++nsub)
#pragma unroll
            for (int ks = 0; ks < 2; ++ks) {
                FragAB bfr;
                const unsigned short* bp = &Kc[(nsub * 16 + l) * HD + ks * 32 + 8 * h];
                bfr.u4[0] = *(const v4u*)(bp);
                bfr.u4[1] = *(const v4u*)(bp + 16);
                sc[nsub] = wmma_bf16(qf[ks], bfr, sc[nsub]);
            }

        // online softmax per row (row m = r + 8h lives in one 16-lane half)
#pragma unroll
        for (int r = 0; r < 8; ++r) {
            float s0 = sc[0][r] * SCALE, s1 = sc[1][r] * SCALE;
            float s2 = sc[2][r] * SCALE, s3 = sc[3][r] * SCALE;
            float mx = fmaxf(fmaxf(s0, s1), fmaxf(s2, s3));
            float mnew = fmaxf(m_i[r], rowmax16(mx));
            float a  = __expf(m_i[r] - mnew);
            float p0 = __expf(s0 - mnew), p1 = __expf(s1 - mnew);
            float p2 = __expf(s2 - mnew), p3 = __expf(s3 - mnew);
            l_i[r] = l_i[r] * a + rowsum16((p0 + p1) + (p2 + p3));
            m_i[r] = mnew;
            alpha[r] = a;
            int pb = (r + 8 * h) * KC + l;            // C layout -> LDS
            Psh[wave][pb]      = f2bf(p0);
            Psh[wave][pb + 16] = f2bf(p1);
            Psh[wave][pb + 32] = f2bf(p2);
            Psh[wave][pb + 48] = f2bf(p3);
        }
#pragma unroll
        for (int j = 0; j < 4; ++j)
#pragma unroll
            for (int r = 0; r < 8; ++r) acc[j][r] *= alpha[r];

        // in-wave LDS ordering: P stores above, A-frag loads below
        asm volatile("s_wait_dscnt 0" ::: "memory");

        // acc += P(16x64) x V(64x64) -> 8 WMMA
#pragma unroll
        for (int kp = 0; kp < 2; ++kp) {
            FragAB pf;
            const unsigned short* pp = &Psh[wave][l * KC + kp * 32];
            pf.u4[0] = *(const v4u*)(pp + 8 * h);
            pf.u4[1] = *(const v4u*)(pp + 16 + 8 * h);
#pragma unroll
            for (int j = 0; j < 4; ++j) {
                FragAB bfr;
                const unsigned short* bp = &Vc[(j * 16 + l) * KC + kp * 32 + 8 * h];
                bfr.u4[0] = *(const v4u*)(bp);
                bfr.u4[1] = *(const v4u*)(bp + 16);
                acc[j] = wmma_bf16(pf, bfr, acc[j]);
            }
        }
    }

    // epilogue: normalize, write bf16 attn-out as [B, S, H*Hd] rows
#pragma unroll
    for (int j = 0; j < 4; ++j)
#pragma unroll
        for (int r = 0; r < 8; ++r) {
            int srow = qRow + r + 8 * h;
            int d    = j * 16 + l;
            float o  = acc[j][r] / l_i[r];
            O[((size_t)(b * S_LEN + srow) * NH + head) * HD + d] = f2bf(o);
        }
}

// ---------------------------------------------------------------------------
// Kernel 3: out = attn @ W_proj + b_proj (f32 output)
// ---------------------------------------------------------------------------
__global__ __launch_bounds__(128) void proj_kernel(
    const unsigned short* __restrict__ A, const float* __restrict__ Wp,
    const float* __restrict__ bp, float* __restrict__ out)
{
    __shared__ unsigned short Ash[64 * 40];
    __shared__ unsigned short Bsh[64 * 40];

    const int tid  = threadIdx.x;
    const int wave = tid >> 5;
    const int lane = tid & 31;
    const int l    = lane & 15;
    const int h    = lane >> 4;
    const int mBase = blockIdx.x * 64;
    const int nBase = blockIdx.y * 64;

    v8f c[4] = {};

    for (int kt = 0; kt < DIM; kt += 32) {
#pragma unroll
        for (int i = 0; i < 16; ++i) {
            int idx = tid + i * 128;
            int r = idx >> 5, cc = idx & 31;
            Ash[r * 40 + cc] = A[(size_t)(mBase + r) * DIM + kt + cc]; // bf16 already
        }
#pragma unroll
        for (int i = 0; i < 16; ++i) {
            int idx = tid + i * 128;
            int kk = idx >> 6, nn = idx & 63;
            Bsh[nn * 40 + kk] = f2bf(Wp[(size_t)(kt + kk) * DIM + nBase + nn]);
        }
        __syncthreads();

        FragAB a;
        const unsigned short* ap = &Ash[(wave * 16 + l) * 40];
        a.u4[0] = *(const v4u*)(ap + 8 * h);
        a.u4[1] = *(const v4u*)(ap + 16 + 8 * h);
#pragma unroll
        for (int j = 0; j < 4; ++j) {
            FragAB bfr;
            const unsigned short* bp2 = &Bsh[(j * 16 + l) * 40 + 8 * h];
            bfr.u4[0] = *(const v4u*)(bp2);
            bfr.u4[1] = *(const v4u*)(bp2 + 16);
            c[j] = wmma_bf16(a, bfr, c[j]);
        }
        __syncthreads();
    }

#pragma unroll
    for (int j = 0; j < 4; ++j) {
        int ncol = nBase + j * 16 + l;
        float bias = bp[ncol];
#pragma unroll
        for (int r = 0; r < 8; ++r) {
            int row = mBase + wave * 16 + r + 8 * h;
            out[(size_t)row * DIM + ncol] = c[j][r] + bias;
        }
    }
}

// ---------------------------------------------------------------------------
extern "C" void kernel_launch(void* const* d_in, const int* in_sizes, int n_in,
                              void* d_out, int out_size, void* d_ws, size_t ws_size,
                              hipStream_t stream)
{
    (void)in_sizes; (void)n_in; (void)out_size; (void)ws_size;

    const float* x    = (const float*)d_in[0];
    const float* Wqkv = (const float*)d_in[1];
    const float* bqkv = (const float*)d_in[2];
    const float* Wp   = (const float*)d_in[3];
    const float* bpj  = (const float*)d_in[4];

    const size_t elems = (size_t)2 * NH * S_LEN * HD;  // 4,194,304 per buffer
    unsigned short* Q  = (unsigned short*)d_ws;
    unsigned short* K  = Q + elems;
    unsigned short* Vt = K + elems;                    // [B,H,Hd,S]
    unsigned short* AO = Vt + elems;                   // [B,S,512] bf16

    qkv_kernel <<<dim3(128, 24),   128, 0, stream>>>(x, Wqkv, bqkv, Q, K, Vt);
    attn_kernel<<<dim3(64, NH, 2), 128, 0, stream>>>(Q, K, Vt, AO);
    proj_kernel<<<dim3(128, 8),    128, 0, stream>>>(AO, Wp, bpj, (float*)d_out);
}
